// Spectral_Attention_Structured_Mesh_2D_84413287235976
// MI455X (gfx1250) — matrix operand order repaired and checked by reading, not compiled, not run on hardware
//
#include <hip/hip_runtime.h>
#include <hip/hip_bf16.h>

// ---------------------------------------------------------------------------
// Spectral attention (Haar DWT + kernelized linear attention) for MI455X.
// GEMMs via v_wmma_f32_16x16x32_f16; fp16 intermediates; A-tiles moved with
// the Tensor Data Mover (tensor_load_to_lds) and async global->LDS copies.
// ---------------------------------------------------------------------------

typedef __attribute__((ext_vector_type(16))) _Float16 v16h;
typedef __attribute__((ext_vector_type(8)))  float    v8f;
typedef unsigned uint32x4 __attribute__((ext_vector_type(4)));
typedef unsigned uint32x8 __attribute__((ext_vector_type(8)));
typedef int v4i __attribute__((ext_vector_type(4)));

union Frag16 { v16h h; unsigned u[8]; };

#define GLOBAL_AS __attribute__((address_space(1)))
#define LDS_AS    __attribute__((address_space(3)))

#ifndef USE_TDM
#define USE_TDM 1
#endif

#if defined(__has_builtin)
#if __has_builtin(__builtin_amdgcn_global_load_async_to_lds_b128)
#define HAVE_ASYNC_LDS 1
#endif
#endif

// Problem constants (shapes fixed by the reference setup_inputs()).
constexpr int Bc  = 8;
constexpr int Hc  = 128;
constexpr int Wc  = 128;
constexpr int Cc  = 256;
constexpr int Nc  = Hc * Wc;          // 16384
constexpr int Hoc = Hc / 2;           // 64
constexpr int Woc = Wc / 2;           // 64
constexpr int Mc  = Hoc * Woc;        // 4096
constexpr int NHc = 8;
constexpr int Dc  = Cc / NHc;         // 32

__device__ __forceinline__ float phi_map(float t) {
    // elu(t) + 1  ==  t>0 ? t+1 : exp(t)
    return t > 0.0f ? t + 1.0f : __expf(t);
}

// ---------------------------------------------------------------------------
// TDM helper: DMA a 2D fp16 tile (tile_h rows x tile_w elems, row stride
// row_elems) from global memory into LDS at lds_byte_addr. 2-group D#
// (tensors up to 2D -> VADDR2/VADDR3 = NULL). Issued per-wave; caller gates
// to a single wave and waits on TENSORcnt.
// ---------------------------------------------------------------------------
__device__ __forceinline__ void tdm_load_tile_f16(const void* gsrc,
                                                  unsigned lds_byte_addr,
                                                  unsigned row_elems,
                                                  unsigned tile_w,
                                                  unsigned tile_h)
{
#if USE_TDM
    const unsigned long long ga = (unsigned long long)gsrc;
    uint32x4 g0;
    g0[0] = 1u;                                   // count=1, user mode
    g0[1] = lds_byte_addr;                        // lds_addr
    g0[2] = (unsigned)ga;                         // global_addr[31:0]
    g0[3] = (unsigned)(ga >> 32) | (2u << 30);    // global_addr[56:32] | type=2
    const unsigned td0 = row_elems;               // tensor_dim0 (elements)
    const unsigned td1 = 0x100000u;               // tensor_dim1 (generous)
    uint32x8 g1;
    g1[0] = (1u << 16);                           // data_size = 1 (2 bytes)
    g1[1] = (td0 & 0xFFFFu) << 16;                // tensor_dim0[15:0]
    g1[2] = (td0 >> 16) | ((td1 & 0xFFFFu) << 16);
    g1[3] = (td1 >> 16) | (tile_w << 16);         // tile_dim0
    g1[4] = tile_h;                               // tile_dim1 (tile_dim2=0)
    g1[5] = row_elems;                            // tensor_dim0_stride[31:0]
    g1[6] = 0u;                                   // stride[47:32] | dim1_stride lo
    g1[7] = 0u;
    asm volatile("tensor_load_to_lds %0, %1" :: "s"(g0), "s"(g1) : "memory");
#else
    (void)gsrc; (void)lds_byte_addr; (void)row_elems; (void)tile_w; (void)tile_h;
#endif
}

__device__ __forceinline__ unsigned lds_addr_of(const void* p) {
    return (unsigned)(unsigned long long)(LDS_AS void*)p;
}

// ---------------------------------------------------------------------------
// Kernel 1: Haar DWT (4 subbands, channel-concat) + exact IDWT branch.
// ---------------------------------------------------------------------------
__global__ __launch_bounds__(256)
void dwt_kernel(const float* __restrict__ x,
                _Float16* __restrict__ subs,   // [B, M, 4C]
                _Float16* __restrict__ idwt)   // [B, N, C]
{
    const int c   = threadIdx.x;
    const int blk = blockIdx.x;                // b*4096 + ho*64 + wo
    const int b   = blk >> 12;
    const int rem = blk & 4095;
    const int ho  = rem >> 6;
    const int wo  = rem & 63;

    const size_t base = ((size_t)b * Nc + (size_t)(2 * ho) * Wc + 2 * wo) * Cc + c;
    const float a  = x[base];
    const float bb = x[base + Cc];
    const float cc = x[base + (size_t)Wc * Cc];
    const float dd = x[base + (size_t)Wc * Cc + Cc];

    const float s0 = 0.5f * (a + bb + cc + dd);
    const float s1 = 0.5f * (a + bb - cc - dd);
    const float s2 = 0.5f * (a - bb + cc - dd);
    const float s3 = 0.5f * (a - bb - cc + dd);

    const size_t m  = (size_t)ho * Woc + wo;
    const size_t sb = ((size_t)b * Mc + m) * (4 * Cc) + c;
    subs[sb]           = (_Float16)s0;
    subs[sb + Cc]      = (_Float16)s1;
    subs[sb + 2 * Cc]  = (_Float16)s2;
    subs[sb + 3 * Cc]  = (_Float16)s3;

    const float y00 = 0.5f * (s0 + s1 + s2 + s3);
    const float y01 = 0.5f * (s0 + s1 - s2 - s3);
    const float y10 = 0.5f * (s0 - s1 + s2 - s3);
    const float y11 = 0.5f * (s0 - s1 - s2 + s3);
    idwt[base]                        = (_Float16)y00;
    idwt[base + Cc]                   = (_Float16)y01;
    idwt[base + (size_t)Wc * Cc]      = (_Float16)y10;
    idwt[base + (size_t)Wc * Cc + Cc] = (_Float16)y11;
}

// ---------------------------------------------------------------------------
// Kernel 2: generic WMMA GEMM   C[M x N] = op(A[M x K] * Bw[K x N])
//   A fp16 tiles are DMA'd by the Tensor Data Mover; fp32 A converts on stage.
//   Block 256 threads = 8 waves; block tile 128x64; wave tile 32x32.
//   EPI: 0 = identity, 1 = phi (elu+1).
// ---------------------------------------------------------------------------
template<typename AT, int EPI, typename OT>
__global__ __launch_bounds__(256)
void gemm_wmma(const AT* __restrict__ A0, const AT* __restrict__ A1, int splitK,
               int lda0, int lda1,
               const float* __restrict__ Bw, int ldb,
               OT* __restrict__ Cmat, int ldc, int K)
{
    constexpr int BM = 128, BN = 64, BK = 32;
    __shared__ _Float16 As[BM][BK];        // row-major A tile (8 KB)
    __shared__ _Float16 Bs[BN][BK];        // transposed B tile Bs[n][k] (4 KB)

    const int t    = threadIdx.x;
    const int lane = t & 31;
    const int wv   = t >> 5;
    const int wr   = wv & 3;               // 4 wave-rows
    const int wc   = wv >> 2;              // 2 wave-cols
    const int hg   = lane >> 4;
    const int l15  = lane & 15;
    const size_t rowBase = (size_t)blockIdx.x * BM;
    const int    colBase = blockIdx.y * BN;

    v8f acc[2][2] = {};

    for (int k0 = 0; k0 < K; k0 += BK) {
        // --- stage A tile ---
        if constexpr (sizeof(AT) == 2 && USE_TDM) {
            // TDM DMA: one wave issues the 2D tile descriptor, waits TENSORcnt
            if (wv == 0) {
                const int inA0 = (k0 < splitK);
                const AT* gtile = inA0
                    ? (A0 + rowBase * (size_t)lda0 + k0)
                    : (A1 + rowBase * (size_t)lda1 + (k0 - splitK));
                tdm_load_tile_f16(gtile, lds_addr_of(&As[0][0]),
                                  (unsigned)(inA0 ? lda0 : lda1), BK, BM);
                __builtin_amdgcn_s_wait_tensorcnt(0);
            }
        } else {
            const int row = t >> 1;
            const int cb  = (t & 1) * 16;
            const size_t grow = rowBase + row;
#pragma unroll
            for (int j = 0; j < 16; ++j) {
                const int kk = k0 + cb + j;
                float v;
                if (kk < splitK) v = (float)A0[grow * (size_t)lda0 + kk];
                else             v = (float)A1[grow * (size_t)lda1 + (kk - splitK)];
                As[row][cb + j] = (_Float16)v;
            }
        }
        // --- stage B tile transposed (f32 -> f16): Bs[n][k] ---
        {
            const int n  = t >> 2;
            const int kb = (t & 3) * 8;
#pragma unroll
            for (int j = 0; j < 8; ++j)
                Bs[n][kb + j] = (_Float16)Bw[(size_t)(k0 + kb + j) * ldb + colBase + n];
        }
        if (k0 + BK < K)
            __builtin_prefetch(&Bw[(size_t)(k0 + BK) * ldb + colBase], 0, 0);
        __syncthreads();

        // --- fragment fills (32-bit LDS reads matching WMMA VGPR layouts) ---
        Frag16 af[2], bf[2];
#pragma unroll
        for (int ti = 0; ti < 2; ++ti)
#pragma unroll
            for (int v = 0; v < 8; ++v) {
                const int kk = 16 * (v >> 2) + 8 * hg + 2 * (v & 3);
                af[ti].u[v] = *(const unsigned*)&As[wr * 32 + ti * 16 + l15][kk];
            }
#pragma unroll
        for (int tj = 0; tj < 2; ++tj)
#pragma unroll
            for (int v = 0; v < 8; ++v) {
                const int kk = 16 * (v >> 2) + 8 * hg + 2 * (v & 3);
                bf[tj].u[v] = *(const unsigned*)&Bs[wc * 32 + tj * 16 + l15][kk];
            }

#pragma unroll
        for (int ti = 0; ti < 2; ++ti)
#pragma unroll
            for (int tj = 0; tj < 2; ++tj)
                acc[ti][tj] = __builtin_amdgcn_wmma_f32_16x16x32_f16(
                    false, af[ti].h, false, bf[tj].h,
                    (short)0, acc[ti][tj], false, false);
        __syncthreads();
    }

    // --- epilogue ---
#pragma unroll
    for (int ti = 0; ti < 2; ++ti)
#pragma unroll
        for (int tj = 0; tj < 2; ++tj)
#pragma unroll
            for (int r = 0; r < 8; ++r) {
                const size_t row = rowBase + wr * 32 + ti * 16 + r + 8 * hg;
                const int    col = colBase + wc * 32 + tj * 16 + l15;
                float v = acc[ti][tj][r];
                if (EPI == 1) v = phi_map(v);
                Cmat[row * (size_t)ldc + col] = (OT)v;
            }
}

// ---------------------------------------------------------------------------
// Kernel 3: per-(b,h) reduction  kvE = [ K^T V | ksum | 0 ]  (32 x 48)
//   M=4096 split over 8 waves; V extended with a ones column so ksum falls
//   out of the same WMMA. Cross-wave reduce via LDS ds_add_f32.
// ---------------------------------------------------------------------------
__global__ __launch_bounds__(256)
void kv_kernel(const _Float16* __restrict__ kbuf,   // [B, M, C]
               const _Float16* __restrict__ vbuf,   // [B, M, C]
               _Float16* __restrict__ kvE)          // [B*h, 32, 48]
{
    const int bh = blockIdx.x;
    const int b  = bh >> 3;
    const int h  = bh & 7;

    __shared__ _Float16 kT[8][32][32];   // per wave: kT[d][m]  (16 KB)
    __shared__ _Float16 vT[8][48][32];   // per wave: vT[e][m]  (24 KB)
    __shared__ float    red[32 * 48];    // 6 KB

    const int t    = threadIdx.x;
    const int wv   = t >> 5;
    const int lane = t & 31;
    const int hg   = lane >> 4;
    const int l15  = lane & 15;

#pragma unroll
    for (int j = 0; j < 6; ++j) red[t * 6 + j] = 0.0f;
#pragma unroll
    for (int e = 32; e < 48; ++e)
        vT[wv][e][lane] = (e == 32) ? (_Float16)1.0f : (_Float16)0.0f;
    __syncthreads();

    v8f acc[2][3] = {};

    union Row { uint4 q[4]; _Float16 hx[32]; };

    for (int it = 0; it < 16; ++it) {
        const int m0 = wv * 512 + it * 32;
        // vectorized global row loads (b128), transposed scatter into LDS
        const _Float16* krow = kbuf + ((size_t)b * Mc + m0 + lane) * Cc + h * Dc;
        const _Float16* vrow = vbuf + ((size_t)b * Mc + m0 + lane) * Cc + h * Dc;
        Row kr, vr;
#pragma unroll
        for (int j = 0; j < 4; ++j) kr.q[j] = ((const uint4*)krow)[j];
#pragma unroll
        for (int j = 0; j < 4; ++j) vr.q[j] = ((const uint4*)vrow)[j];
#pragma unroll
        for (int d = 0; d < 32; ++d) kT[wv][d][lane] = kr.hx[d];
#pragma unroll
        for (int e = 0; e < 32; ++e) vT[wv][e][lane] = vr.hx[e];
        __syncthreads();

        Frag16 af[2], bf[3];
#pragma unroll
        for (int ti = 0; ti < 2; ++ti)
#pragma unroll
            for (int v = 0; v < 8; ++v) {
                const int kk = 16 * (v >> 2) + 8 * hg + 2 * (v & 3);
                af[ti].u[v] = *(const unsigned*)&kT[wv][ti * 16 + l15][kk];
            }
#pragma unroll
        for (int tj = 0; tj < 3; ++tj)
#pragma unroll
            for (int v = 0; v < 8; ++v) {
                const int kk = 16 * (v >> 2) + 8 * hg + 2 * (v & 3);
                bf[tj].u[v] = *(const unsigned*)&vT[wv][tj * 16 + l15][kk];
            }
#pragma unroll
        for (int ti = 0; ti < 2; ++ti)
#pragma unroll
            for (int tj = 0; tj < 3; ++tj)
                acc[ti][tj] = __builtin_amdgcn_wmma_f32_16x16x32_f16(
                    false, af[ti].h, false, bf[tj].h,
                    (short)0, acc[ti][tj], false, false);
        __syncthreads();
    }

#pragma unroll
    for (int ti = 0; ti < 2; ++ti)
#pragma unroll
        for (int tj = 0; tj < 3; ++tj)
#pragma unroll
            for (int r = 0; r < 8; ++r) {
                const int row = ti * 16 + r + 8 * hg;
                const int col = tj * 16 + l15;
                atomicAdd(&red[row * 48 + col], acc[ti][tj][r]);
            }
    __syncthreads();

#pragma unroll
    for (int j = 0; j < 6; ++j) {
        const int idx = t * 6 + j;
        kvE[(size_t)bh * 1536 + idx] = (_Float16)red[idx];
    }
}

// ---------------------------------------------------------------------------
// Kernel 4: attn = (qf @ kvE) * z, column 32 of kvE supplies the denominator.
//   qf tile (256 rows x 32) staged to LDS with async global->LDS b128 copies.
// ---------------------------------------------------------------------------
__global__ __launch_bounds__(256)
void attn_kernel(const _Float16* __restrict__ qf,    // [B, N, C]
                 const _Float16* __restrict__ kvE,   // [B*h, 32, 48]
                 _Float16* __restrict__ attn)        // [B, N, C]
{
    const int bh = blockIdx.x;
    const int b  = bh >> 3;
    const int h  = bh & 7;

    __shared__ _Float16 Bs[48][32];        // transposed kvE: Bs[e][d] (3 KB)
    __shared__ _Float16 Aq[256 * 32];      // qf tile, row-major (16 KB)

    const int t    = threadIdx.x;
    const int wv   = t >> 5;
    const int lane = t & 31;
    const int hg   = lane >> 4;
    const int l15  = lane & 15;

    const size_t rowBlock = (size_t)blockIdx.y * 256;
    const _Float16* gq = qf + ((size_t)b * Nc + rowBlock) * Cc + h * Dc;

    // stage qf tile: 256 rows x 32 halfs = 1024 x 16B chunks, 4 per thread
#if defined(HAVE_ASYNC_LDS)
#pragma unroll
    for (int j = 0; j < 4; ++j) {
        const int id  = j * 256 + t;
        const int row = id >> 2, seg = id & 3;
        __builtin_amdgcn_global_load_async_to_lds_b128(
            (GLOBAL_AS v4i*)(gq + (size_t)row * Cc + seg * 8),
            (LDS_AS v4i*)&Aq[row * 32 + seg * 8], 0, 0);
    }
    asm volatile("s_wait_asynccnt 0x0" ::: "memory");
#else
#pragma unroll
    for (int j = 0; j < 4; ++j) {
        const int id  = j * 256 + t;
        const int row = id >> 2, seg = id & 3;
        *(uint4*)&Aq[row * 32 + seg * 8] =
            *(const uint4*)(gq + (size_t)row * Cc + seg * 8);
    }
#endif

    // stage kvE transposed: 1536 elements, 6 per thread
#pragma unroll
    for (int j = 0; j < 6; ++j) {
        const int idx = t * 6 + j;
        const int n = idx >> 5, k = idx & 31;
        Bs[n][k] = kvE[(size_t)bh * 1536 + (size_t)k * 48 + n];
    }
    __syncthreads();

    Frag16 af[2], bf[3];
#pragma unroll
    for (int ti = 0; ti < 2; ++ti)
#pragma unroll
        for (int v = 0; v < 8; ++v) {
            const int kk = 16 * (v >> 2) + 8 * hg + 2 * (v & 3);
            af[ti].u[v] =
                *(const unsigned*)&Aq[(wv * 32 + ti * 16 + l15) * 32 + kk];
        }
#pragma unroll
    for (int tj = 0; tj < 3; ++tj)
#pragma unroll
        for (int v = 0; v < 8; ++v) {
            const int kk = 16 * (v >> 2) + 8 * hg + 2 * (v & 3);
            bf[tj].u[v] = *(const unsigned*)&Bs[tj * 16 + l15][kk];
        }

    v8f acc[2][3] = {};
#pragma unroll
    for (int ti = 0; ti < 2; ++ti)
#pragma unroll
        for (int tj = 0; tj < 3; ++tj)
            acc[ti][tj] = __builtin_amdgcn_wmma_f32_16x16x32_f16(
                false, af[ti].h, false, bf[tj].h,
                (short)0, acc[ti][tj], false, false);

    // normalize: z = 1/(col32 + 1e-6), broadcast within each 16-lane group
#pragma unroll
    for (int ti = 0; ti < 2; ++ti) {
        v8f dv = acc[ti][2];
#pragma unroll
        for (int r = 0; r < 8; ++r) {
            const float den = __shfl(dv[r], lane & 16, 32);
            const float rz  = 1.0f / (den + 1e-6f);
            const size_t row = rowBlock + wv * 32 + ti * 16 + r + 8 * hg;
#pragma unroll
            for (int tj = 0; tj < 2; ++tj) {
                const int col = tj * 16 + l15;
                attn[((size_t)b * Nc + row) * Cc + h * Dc + col] =
                    (_Float16)(acc[ti][tj][r] * rz);
            }
        }
    }
}

// ---------------------------------------------------------------------------
// Host orchestration
// ---------------------------------------------------------------------------
extern "C" void kernel_launch(void* const* d_in, const int* in_sizes, int n_in,
                              void* d_out, int out_size, void* d_ws, size_t ws_size,
                              hipStream_t stream) {
    const float* x     = (const float*)d_in[0];
    // d_in[1], d_in[2]: Haar dec/rec filters (fixed; hardcoded in dwt_kernel)
    const float* w_q   = (const float*)d_in[3];
    const float* w_k   = (const float*)d_in[4];
    const float* w_v   = (const float*)d_in[5];
    const float* w_red = (const float*)d_in[6];
    const float* w_o   = (const float*)d_in[7];
    float* out = (float*)d_out;

    char* p = (char*)d_ws;
    _Float16* subs = (_Float16*)p; p += (size_t)Bc * Mc * 4 * Cc * 2;   // 64 MB
    _Float16* idwt = (_Float16*)p; p += (size_t)Bc * Nc * Cc * 2;       // 64 MB
    _Float16* qf   = (_Float16*)p; p += (size_t)Bc * Nc * Cc * 2;       // 64 MB
    _Float16* kvin = (_Float16*)p; p += (size_t)Bc * Mc * Cc * 2;       // 16 MB
    _Float16* kbuf = (_Float16*)p; p += (size_t)Bc * Mc * Cc * 2;       // 16 MB
    _Float16* vbuf = (_Float16*)p; p += (size_t)Bc * Mc * Cc * 2;       // 16 MB
    _Float16* attn = (_Float16*)p; p += (size_t)Bc * Nc * Cc * 2;       // 64 MB
    _Float16* kvE  = (_Float16*)p; p += (size_t)Bc * NHc * 32 * 48 * 2; // 192 KB

    const int BIG = 1 << 30;

    // 1) DWT + IDWT
    dwt_kernel<<<Bc * Mc, 256, 0, stream>>>(x, subs, idwt);

    // 2) qf = phi(x @ w_q)            [131072 x 256], K=256 (fp32 A path)
    gemm_wmma<float, 1, _Float16><<<dim3(1024, 4), 256, 0, stream>>>(
        x, x, BIG, Cc, Cc, w_q, Cc, qf, Cc, Cc);

    // 3) kv_in = subs @ w_red         [32768 x 256], K=1024 (TDM A path)
    gemm_wmma<_Float16, 0, _Float16><<<dim3(256, 4), 256, 0, stream>>>(
        subs, subs, BIG, 4 * Cc, 4 * Cc, w_red, Cc, kvin, Cc, 4 * Cc);

    // 4) k = phi(kv_in @ w_k),  v = kv_in @ w_v
    gemm_wmma<_Float16, 1, _Float16><<<dim3(256, 4), 256, 0, stream>>>(
        kvin, kvin, BIG, Cc, Cc, w_k, Cc, kbuf, Cc, Cc);
    gemm_wmma<_Float16, 0, _Float16><<<dim3(256, 4), 256, 0, stream>>>(
        kvin, kvin, BIG, Cc, Cc, w_v, Cc, vbuf, Cc, Cc);

    // 5) kvE = [K^T V | ksum] per (b,h)
    kv_kernel<<<Bc * NHc, 256, 0, stream>>>(kbuf, vbuf, kvE);

    // 6) attn = (qf @ kv) * z
    attn_kernel<<<dim3(Bc * NHc, Nc / 256), 256, 0, stream>>>(qf, kvE, attn);

    // 7) out = [attn | idwt] @ w_o    [131072 x 256], K=512 (concat TDM path)
    gemm_wmma<_Float16, 0, float><<<dim3(1024, 4), 256, 0, stream>>>(
        attn, idwt, Cc, Cc, Cc, w_o, Cc, out, Cc, 2 * Cc);
}
